// RowInteraction_51934744543738
// MI455X (gfx1250) — matrix operand order repaired and checked by reading, not compiled, not run on hardware
//
#include <hip/hip_runtime.h>
#include <hip/hip_bf16.h>
#include <math.h>
#include <stdint.h>

typedef _Float16 h16;
typedef __attribute__((ext_vector_type(16))) _Float16 v16h;
typedef __attribute__((ext_vector_type(8)))  float    v8f;
typedef __attribute__((ext_vector_type(4)))  unsigned int u32x4;
typedef __attribute__((ext_vector_type(8)))  int          i32x8;
typedef __attribute__((ext_vector_type(4)))  int          i32x4;

#if defined(__has_builtin)
#  if __has_builtin(__builtin_amdgcn_tensor_load_to_lds)
#    define HAVE_TDM 1
#  endif
#endif
#ifndef HAVE_TDM
#  define HAVE_TDM 0
#endif

namespace {

constexpr int kE    = 128;   // embed dim
constexpr int kNH   = 4;     // heads
constexpr int kHD   = 32;    // head dim
constexpr int kDFF  = 512;   // ffn hidden
constexpr int kNB   = 3;     // blocks
constexpr int kC    = 4;     // cls rows
constexpr int kS    = 104;   // live sequence rows
constexpr int kSP   = 128;   // padded rows (multiple of 32 for WMMA K-dim)
constexpr int kLD32 = 132;   // f32 LDS row stride (bank-conflict pad)
constexpr int kLD16 = 136;   // f16 LDS row stride

// swizzled-weight region bases (units of 512-half tiles)
constexpr int kTilesQKV = 4 * 24;   // K=128/32, N=384/16
constexpr int kTilesWO  = 4 * 8;
constexpr int kTilesF1  = 4 * 32;
constexpr int kTilesF2  = 16 * 8;   // K=512/32, N=128/16
constexpr int kBaseQKV = 0;
constexpr int kBaseWO  = kBaseQKV + kNB * kTilesQKV;  // 288
constexpr int kBaseF1  = kBaseWO  + kNB * kTilesWO;   // 384
constexpr int kBaseF2  = kBaseF1  + kNB * kTilesF1;   // 768
constexpr int kTilesTotal = kBaseF2 + kNB * kTilesF2; // 1152
constexpr size_t kWsHalves = (size_t)kTilesTotal * 512;

// ---- WMMA fragment loaders (gfx1250 wave32 layouts, CDNA5 ISA) ----

// A 16x32 f16: lane%16 = M row; K = h + (lane>=16 ? 16 : 0)
__device__ __forceinline__ v16h frag_a(const h16* A, int row0, int k0, int m, int hi) {
  const h16* p = A + (row0 + m) * kLD16 + k0 + hi * 16;
  v16h a;
#pragma unroll
  for (int h = 0; h < 8; ++h) a[h] = p[h];
#pragma unroll
  for (int h = 0; h < 8; ++h) a[h + 8] = p[8 + h];
  return a;
}

// B 32x16 from a row-major matrix used transposed: B[k][n] = Bm[n0+n][k0+k]
// (contiguous 16-half run per lane -> two ds_load_b128)
__device__ __forceinline__ v16h frag_bT(const h16* Bm, int k0, int n0, int n, int hi) {
  const h16* p = Bm + (n0 + n) * kLD16 + k0 + hi * 16;
  v16h b;
#pragma unroll
  for (int h = 0; h < 16; ++h) b[h] = p[h];
  return b;
}

// B from global f32 weights, converted on load (fallback path)
__device__ __forceinline__ v16h frag_bg(const float* B, int ldb, int k0, int n0, int n, int hi) {
  const float* p = B + (size_t)(k0 + hi * 16) * ldb + n0 + n;
  v16h b;
#pragma unroll
  for (int h = 0; h < 16; ++h) b[h] = (h16)p[(size_t)h * ldb];
  return b;
}

// B from pre-swizzled f16 weights: one 32-byte contiguous load per lane
__device__ __forceinline__ v16h frag_bw(const h16* Wsw, int ntn, int k0, int n0, int lane) {
  const size_t tile = (size_t)(k0 >> 5) * ntn + (n0 >> 4);
  return *(const v16h*)(Wsw + tile * 512 + (size_t)lane * 16);
}

__device__ __forceinline__ v8f wmma_f16(v16h a, v16h b, v8f c) {
  return __builtin_amdgcn_wmma_f32_16x16x32_f16(false, a, false, b, (short)0, c, false, false);
}

// LayerNorm rows of X (f32) -> Y (f16); zero-fill pad rows. One row per thread.
__device__ __forceinline__ void layernorm_to_f16(const float* X, h16* Y,
                                                 const float* g, const float* bta, int tid) {
  if (tid < kSP) {
    const int s = tid;
    h16* y = Y + s * kLD16;
    if (s < kS) {
      const float* row = X + s * kLD32;
      float m = 0.f;
#pragma unroll 4
      for (int c = 0; c < kE; ++c) m += row[c];
      m *= (1.f / kE);
      float v = 0.f;
#pragma unroll 4
      for (int c = 0; c < kE; ++c) { float d = row[c] - m; v += d * d; }
      v *= (1.f / kE);
      const float rs = rsqrtf(v + 1e-5f);
#pragma unroll 4
      for (int c = 0; c < kE; ++c) y[c] = (h16)((row[c] - m) * rs * g[c] + bta[c]);
    } else {
#pragma unroll 4
      for (int c = 0; c < kE; ++c) y[c] = (h16)0.f;
    }
  }
}

#if HAVE_TDM
// TDM: DMA rows 4..103 of this sequence (100 x 128 f32) into X rows 4..103.
// pad_enable: +4 DWORDs after every 128 DWORDs -> LDS row stride 132 floats.
__device__ __forceinline__ void tdm_load_seq(const float* src_row4, float* x_row4) {
  const unsigned long long ga = (unsigned long long)(uintptr_t)src_row4;
  const unsigned lds_off = (unsigned)(uintptr_t)x_row4;  // low 32 bits = LDS byte offset
  u32x4 g0;
  g0[0] = 1u;                                        // count=1, user mode, no gather
  g0[1] = lds_off;                                   // lds_addr
  g0[2] = (unsigned)(ga & 0xFFFFFFFFu);              // global_addr[31:0]
  g0[3] = (unsigned)((ga >> 32) & 0x01FFFFFFu)       // global_addr[56:32]
        | (2u << 30);                                // type = 2 ("image")
  i32x8 g1;
  g1[0] = (int)((2u << 16)                           // data_size = 4 bytes
        | (1u << 20)                                 // pad_enable
        | (6u << 22)                                 // pad_interval: 128 DWORDs
        | (3u << 25));                               // pad_amount: 4 DWORDs
  g1[1] = (int)(128u << 16);                         // tensor_dim0[15:0] = 128
  g1[2] = (int)(100u << 16);                         // dim0 hi=0 | tensor_dim1[15:0] = 100
  g1[3] = (int)(128u << 16);                         // dim1 hi=0 | tile_dim0 = 128
  g1[4] = (int)(100u);                               // tile_dim1 = 100, tile_dim2 = 0
  g1[5] = (int)(128u);                               // tensor_dim0_stride = 128
  g1[6] = 0;
  g1[7] = 0;
  const i32x4 z4 = {0, 0, 0, 0};
#if __has_include(<hip/amd_detail/amd_gfx1250_TDM.h>)
  const i32x8 z8 = {0, 0, 0, 0, 0, 0, 0, 0};
  __builtin_amdgcn_tensor_load_to_lds(g0, g1, z4, z4, z8, 0);
#else
  __builtin_amdgcn_tensor_load_to_lds(g0, g1, z4, z4, 0);
#endif
}
#endif

} // namespace

// one-time pre-pass: convert all weights to f16 pre-swizzled into WMMA B-fragment order
__global__ void swizzle_weights_gfx1250(const float* __restrict__ w_qkv,
                                        const float* __restrict__ w_o,
                                        const float* __restrict__ w1,
                                        const float* __restrict__ w2,
                                        h16* __restrict__ ws) {
  const int gid = blockIdx.x * blockDim.x + threadIdx.x;
  if (gid >= kTilesTotal * 32) return;
  const int lane = gid & 31;
  const int tile = gid >> 5;
  const int n = lane & 15, hi = lane >> 4;

  const float* W; int ldb, ntn, t;
  if (tile < kBaseWO) {
    const int j = tile - kBaseQKV, b = j / kTilesQKV; t = j % kTilesQKV;
    W = w_qkv + (size_t)b * kE * 3 * kE; ldb = 3 * kE; ntn = 24;
  } else if (tile < kBaseF1) {
    const int j = tile - kBaseWO, b = j / kTilesWO; t = j % kTilesWO;
    W = w_o + (size_t)b * kE * kE; ldb = kE; ntn = 8;
  } else if (tile < kBaseF2) {
    const int j = tile - kBaseF1, b = j / kTilesF1; t = j % kTilesF1;
    W = w1 + (size_t)b * kE * kDFF; ldb = kDFF; ntn = 32;
  } else {
    const int j = tile - kBaseF2, b = j / kTilesF2; t = j % kTilesF2;
    W = w2 + (size_t)b * kDFF * kE; ldb = kE; ntn = 8;
  }
  const int k0 = (t / ntn) * 32, n0 = (t % ntn) * 16;
  const float* p = W + (size_t)(k0 + hi * 16) * ldb + n0 + n;
  h16* dst = ws + (size_t)tile * 512 + (size_t)lane * 16;
#pragma unroll
  for (int h = 0; h < 16; ++h) dst[h] = (h16)p[(size_t)h * ldb];
}

template <bool USE_WS>
__global__ __launch_bounds__(256, 1)
void fused_cls_transformer_gfx1250(const float* __restrict__ emb,
                                   const float* __restrict__ cls,
                                   const float* __restrict__ ln1_g, const float* __restrict__ ln1_b,
                                   const float* __restrict__ w_qkv, const float* __restrict__ b_qkv,
                                   const float* __restrict__ w_o,   const float* __restrict__ b_o,
                                   const float* __restrict__ ln2_g, const float* __restrict__ ln2_b,
                                   const float* __restrict__ w1,    const float* __restrict__ b1,
                                   const float* __restrict__ w2,    const float* __restrict__ b2,
                                   const float* __restrict__ og,    const float* __restrict__ ob,
                                   const h16* __restrict__ ws,
                                   float* __restrict__ out)
{
  extern __shared__ char smem[];
  float* X  = (float*)smem;                 // residual stream, kSP x kLD32 (f32)
  float* S  = X + kSP * kLD32;              // attention scores,  kSP x kLD32 (f32)
  h16* XN = (h16*)(S + kSP * kLD32);        // LN1 out; reused as attention O
  h16* Qb = XN + kSP * kLD16;               // Q; reused as LN2 out
  h16* Kb = Qb + kSP * kLD16;               // K
  h16* Vt = Kb + kSP * kLD16;               // V stored TRANSPOSED: [dim][seq]
  h16* Pb = Vt + kSP * kLD16;               // softmax probs; reused as FFN hidden chunk

  const int tid  = threadIdx.x;
  const int lane = tid & 31;
  const int wave = tid >> 5;
  const int n    = lane & 15;
  const int hi   = lane >> 4;
  const int bt   = blockIdx.x;  // (b*T + t)

  const float* src = emb + (size_t)bt * kS * kE;

#if HAVE_TDM
  // rows 4..103 via Tensor Data Mover; rows 0..3 (cls) and 104..127 (zero) by VALU
  if (wave == 0) {
    tdm_load_seq(src + (size_t)kC * kE, X + kC * kLD32);
  }
  for (int idx = tid; idx < (kC + (kSP - kS)) * kE; idx += 256) {
    const int r = idx >> 7, c = idx & (kE - 1);
    if (r < kC) X[r * kLD32 + c] = cls[r * kE + c];
    else        X[(kS + r - kC) * kLD32 + c] = 0.f;
  }
  if (wave == 0) __builtin_amdgcn_s_wait_tensorcnt(0);
#else
  for (int idx = tid; idx < kSP * kE; idx += 256) {
    const int s = idx >> 7, c = idx & (kE - 1);
    float v = 0.f;
    if (s < kC)      v = cls[s * kE + c];
    else if (s < kS) v = src[s * kE + c];
    X[s * kLD32 + c] = v;
  }
#endif
  __syncthreads();

  for (int blk = 0; blk < kNB; ++blk) {
    const float* wqkv = w_qkv + (size_t)blk * kE * 3 * kE;
    const float* bqkv = b_qkv + blk * 3 * kE;
    const float* wo   = w_o   + (size_t)blk * kE * kE;
    const float* bo   = b_o   + blk * kE;
    const float* wf1  = w1    + (size_t)blk * kE * kDFF;
    const float* bf1  = b1    + blk * kDFF;
    const float* wf2  = w2    + (size_t)blk * kDFF * kE;
    const float* bf2  = b2    + blk * kE;
    const h16* qkv_sw = ws + (size_t)(kBaseQKV + blk * kTilesQKV) * 512;
    const h16* wo_sw  = ws + (size_t)(kBaseWO  + blk * kTilesWO ) * 512;
    const h16* wf1_sw = ws + (size_t)(kBaseF1  + blk * kTilesF1 ) * 512;
    const h16* wf2_sw = ws + (size_t)(kBaseF2  + blk * kTilesF2 ) * 512;

    // ---- LN1 -> XN ----
    layernorm_to_f16(X, XN, ln1_g + blk * kE, ln1_b + blk * kE, tid);
    __syncthreads();

    // ---- QKV: [128x128] @ [128x384] + bias ----
    for (int t = wave; t < 8 * 24; t += 8) {
      const int row0 = (t / 24) * 16;
      const int col0 = (t % 24) * 16;
      v8f acc;
      const float bv = bqkv[col0 + n];
#pragma unroll
      for (int r = 0; r < 8; ++r) acc[r] = bv;
#pragma unroll
      for (int k0 = 0; k0 < kE; k0 += 32) {
        v16h a = frag_a(XN, row0, k0, n, hi);
        v16h b;
        if constexpr (USE_WS) b = frag_bw(qkv_sw, 24, k0, col0, lane);
        else                  b = frag_bg(wqkv, 3 * kE, k0, col0, n, hi);
        acc = wmma_f16(a, b, acc);
      }
      if (col0 < 2 * kE) {  // Q, K: row-major
        h16* dst = (col0 < kE) ? Qb : Kb;
        const int cc = (col0 & (kE - 1)) + n;
#pragma unroll
        for (int r = 0; r < 8; ++r) dst[(row0 + hi * 8 + r) * kLD16 + cc] = (h16)acc[r];
      } else {              // V: transposed [dim][seq], contiguous 8-half store
        h16* dst = Vt + (col0 - 2 * kE + n) * kLD16 + row0 + hi * 8;
#pragma unroll
        for (int r = 0; r < 8; ++r) dst[r] = (h16)acc[r];
      }
    }
    __syncthreads();

    // ---- RoPE on Q and K ----
    for (int idx = tid; idx < kS * 64; idx += 256) {
      const int s = idx >> 6;
      const int p = idx & 63;
      const int head = p >> 4;
      const int j = p & 15;
      const float inv = __expf(-(float)j * (11.512925465f / 16.f)); // base^(-2j/32)
      float sn, cs;
      __sincosf((float)s * inv, &sn, &cs);
      const int c0 = s * kLD16 + head * kHD + 2 * j;
      {
        const float x1 = (float)Qb[c0], x2 = (float)Qb[c0 + 1];
        Qb[c0]     = (h16)(x1 * cs - x2 * sn);
        Qb[c0 + 1] = (h16)(x1 * sn + x2 * cs);
      }
      {
        const float x1 = (float)Kb[c0], x2 = (float)Kb[c0 + 1];
        Kb[c0]     = (h16)(x1 * cs - x2 * sn);
        Kb[c0 + 1] = (h16)(x1 * sn + x2 * cs);
      }
    }
    __syncthreads();

    // ---- attention, head by head ----
    for (int head = 0; head < kNH; ++head) {
      // scores = (Q_h @ K_h^T) / sqrt(HD)   (single K=32 WMMA)
      for (int t = wave; t < 64; t += 8) {
        const int row0 = (t >> 3) * 16, col0 = (t & 7) * 16;
        v8f acc;
#pragma unroll
        for (int r = 0; r < 8; ++r) acc[r] = 0.f;
        v16h a = frag_a(Qb, row0, head * kHD, n, hi);
        v16h b = frag_bT(Kb, head * kHD, col0, n, hi);
        acc = wmma_f16(a, b, acc);
#pragma unroll
        for (int r = 0; r < 8; ++r)
          S[(row0 + hi * 8 + r) * kLD32 + col0 + n] = acc[r] * 0.17677669529663687f;
      }
      __syncthreads();
      // row softmax over live k<104 -> Pb (f16), zero tail
      if (tid < kSP) {
        const int s = tid;
        h16* prow = Pb + s * kLD16;
        if (s < kS) {
          const float* srow = S + s * kLD32;
          float m = -1e30f;
          for (int k = 0; k < kS; ++k) m = fmaxf(m, srow[k]);
          float sum = 0.f;
          for (int k = 0; k < kS; ++k) sum += __expf(srow[k] - m);
          const float rs = 1.f / sum;
          for (int k = 0; k < kS; ++k) prow[k] = (h16)(__expf(srow[k] - m) * rs);
          for (int k = kS; k < kSP; ++k) prow[k] = (h16)0.f;
        } else {
          for (int k = 0; k < kSP; ++k) prow[k] = (h16)0.f;
        }
      }
      __syncthreads();
      // O_h = P @ V_h -> column slice of XN; V read via transposed layout (contiguous)
      for (int t = wave; t < 16; t += 8) {
        const int row0 = (t >> 1) * 16;
        const int col0 = head * kHD + (t & 1) * 16;
        v8f acc;
#pragma unroll
        for (int r = 0; r < 8; ++r) acc[r] = 0.f;
#pragma unroll
        for (int k0 = 0; k0 < kSP; k0 += 32) {
          v16h a = frag_a(Pb, row0, k0, n, hi);
          v16h b = frag_bT(Vt, k0, col0, n, hi);  // B[k][n] = Vt[col][k]
          acc = wmma_f16(a, b, acc);
        }
#pragma unroll
        for (int r = 0; r < 8; ++r)
          XN[(row0 + hi * 8 + r) * kLD16 + col0 + n] = (h16)acc[r];
      }
      __syncthreads();
    }

    // ---- O projection + bias + residual into X ----
    for (int t = wave; t < 64; t += 8) {
      const int row0 = (t >> 3) * 16, col0 = (t & 7) * 16;
      v8f acc;
      const float bv = bo[col0 + n];
#pragma unroll
      for (int r = 0; r < 8; ++r) acc[r] = bv;
#pragma unroll
      for (int k0 = 0; k0 < kE; k0 += 32) {
        v16h a = frag_a(XN, row0, k0, n, hi);
        v16h b;
        if constexpr (USE_WS) b = frag_bw(wo_sw, 8, k0, col0, lane);
        else                  b = frag_bg(wo, kE, k0, col0, n, hi);
        acc = wmma_f16(a, b, acc);
      }
#pragma unroll
      for (int r = 0; r < 8; ++r)
        X[(row0 + hi * 8 + r) * kLD32 + col0 + n] += acc[r];
    }
    __syncthreads();

    // ---- LN2 -> Qb ----
    layernorm_to_f16(X, Qb, ln2_g + blk * kE, ln2_b + blk * kE, tid);
    __syncthreads();

    // ---- FFN ----
    v8f facc[8];
#pragma unroll
    for (int i = 0; i < 8; ++i) {
      const int col0 = ((wave + 8 * i) & 7) * 16;
      const float bv = bf2[col0 + n];
#pragma unroll
      for (int r = 0; r < 8; ++r) facc[i][r] = bv;
    }
    for (int chunk = 0; chunk < kDFF / kE; ++chunk) {
      // hidden chunk: gelu(XN2 @ W1[:, chunk] + b1) -> Pb
      for (int t = wave; t < 64; t += 8) {
        const int row0 = (t >> 3) * 16, col0 = (t & 7) * 16;
        v8f acc;
        const float bv = bf1[chunk * kE + col0 + n];
#pragma unroll
        for (int r = 0; r < 8; ++r) acc[r] = bv;
#pragma unroll
        for (int k0 = 0; k0 < kE; k0 += 32) {
          v16h a = frag_a(Qb, row0, k0, n, hi);
          v16h b;
          if constexpr (USE_WS) b = frag_bw(wf1_sw, 32, k0, chunk * kE + col0, lane);
          else                  b = frag_bg(wf1, kDFF, k0, chunk * kE + col0, n, hi);
          acc = wmma_f16(a, b, acc);
        }
#pragma unroll
        for (int r = 0; r < 8; ++r) {
          const float x = acc[r];
          const float g = 0.5f * x * (1.f + erff(x * 0.7071067811865475f)); // exact GELU
          Pb[(row0 + hi * 8 + r) * kLD16 + col0 + n] = (h16)g;
        }
      }
      __syncthreads();
      // FF2: facc += H_chunk @ W2[chunk, :]
#pragma unroll
      for (int i = 0; i < 8; ++i) {
        const int t = wave + 8 * i;
        const int row0 = (t >> 3) * 16, col0 = (t & 7) * 16;
#pragma unroll
        for (int k0 = 0; k0 < kE; k0 += 32) {
          v16h a = frag_a(Pb, row0, k0, n, hi);
          v16h b;
          if constexpr (USE_WS) b = frag_bw(wf2_sw, 8, chunk * kE + k0, col0, lane);
          else                  b = frag_bg(wf2, kE, chunk * kE + k0, col0, n, hi);
          facc[i] = wmma_f16(a, b, facc[i]);
        }
      }
      __syncthreads();
    }
    // FFN residual add
#pragma unroll
    for (int i = 0; i < 8; ++i) {
      const int t = wave + 8 * i;
      const int row0 = (t >> 3) * 16, col0 = (t & 7) * 16;
#pragma unroll
      for (int r = 0; r < 8; ++r)
        X[(row0 + hi * 8 + r) * kLD32 + col0 + n] += facc[i][r];
    }
    __syncthreads();
  }

  // ---- final LayerNorm on cls rows 0..3 -> output (B,T,C*E) ----
  if (tid < kC) {
    const int s = tid;
    const float* row = X + s * kLD32;
    float m = 0.f;
    for (int c = 0; c < kE; ++c) m += row[c];
    m *= (1.f / kE);
    float v = 0.f;
    for (int c = 0; c < kE; ++c) { float d = row[c] - m; v += d * d; }
    v *= (1.f / kE);
    const float rs = rsqrtf(v + 1e-5f);
    float* dst = out + (size_t)bt * (kC * kE) + s * kE;
    for (int c = 0; c < kE; ++c)
      dst[c] = (row[c] - m) * rs * og[c] + ob[c];
  }
}

extern "C" void kernel_launch(void* const* d_in, const int* in_sizes, int n_in,
                              void* d_out, int out_size, void* d_ws, size_t ws_size,
                              hipStream_t stream) {
  (void)n_in; (void)out_size;
  const float* emb   = (const float*)d_in[0];
  const float* cls   = (const float*)d_in[1];
  const float* ln1_g = (const float*)d_in[2];
  const float* ln1_b = (const float*)d_in[3];
  const float* w_qkv = (const float*)d_in[4];
  const float* b_qkv = (const float*)d_in[5];
  const float* w_o   = (const float*)d_in[6];
  const float* b_o   = (const float*)d_in[7];
  const float* ln2_g = (const float*)d_in[8];
  const float* ln2_b = (const float*)d_in[9];
  const float* w_ff1 = (const float*)d_in[10];
  const float* b_ff1 = (const float*)d_in[11];
  const float* w_ff2 = (const float*)d_in[12];
  const float* b_ff2 = (const float*)d_in[13];
  const float* og    = (const float*)d_in[14];
  const float* ob    = (const float*)d_in[15];
  float* out = (float*)d_out;
  h16* ws = (h16*)d_ws;

  const int BT = in_sizes[0] / (kS * kE);  // B*T = 2048
  const size_t smem = (size_t)(2 * kSP * kLD32) * sizeof(float)
                    + (size_t)(5 * kSP * kLD16) * sizeof(h16);   // 309,248 B <= 320 KB
  const bool use_ws = (ws_size >= kWsHalves * sizeof(h16));

  if (use_ws) {
    const int thr = kTilesTotal * 32;  // 36,864
    swizzle_weights_gfx1250<<<dim3((thr + 255) / 256), dim3(256), 0, stream>>>(
        w_qkv, w_o, w_ff1, w_ff2, ws);
    fused_cls_transformer_gfx1250<true><<<dim3(BT), dim3(256), smem, stream>>>(
        emb, cls, ln1_g, ln1_b, w_qkv, b_qkv, w_o, b_o,
        ln2_g, ln2_b, w_ff1, b_ff1, w_ff2, b_ff2, og, ob, ws, out);
  } else {
    fused_cls_transformer_gfx1250<false><<<dim3(BT), dim3(256), smem, stream>>>(
        emb, cls, ln1_g, ln1_b, w_qkv, b_qkv, w_o, b_o,
        ln2_g, ln2_b, w_ff1, b_ff1, w_ff2, b_ff2, og, ob, ws, out);
  }
}
